// RefinementDecoder_23347442221685
// MI455X (gfx1250) — compile-verified, matrix operand order checked by reading, not run on hardware
//
#include <hip/hip_runtime.h>
#include <hip/hip_bf16.h>

#define HD 512
#define NB 512
#define KM 6
#define TS 60
#define OD 2

typedef __attribute__((ext_vector_type(16))) __bf16 v16bf;
typedef __attribute__((ext_vector_type(8)))  float  v8f;
typedef __attribute__((ext_vector_type(4)))  unsigned int v4u;

union bfpair { unsigned int u; __bf16 h[2]; };
union frag_cvt { struct { v4u lo, hi; } p; v16bf v; };

__device__ __forceinline__ float gelu_f(float x) {
    return 0.5f * x * (1.0f + erff(x * 0.70710678118654752f));
}

// Fragment load when this lane's 32 K-values are CONTIGUOUS in LDS at p32.
// ISA 7.12.2 16-bit layout: lanes<16 need K[0..7],K[16..23]; lanes>=16 need
// K[8..15],K[24..31] -> exactly two aligned 16-byte chunks => 2x ds_load_b128.
__device__ __forceinline__ v16bf frag_ld32(const __bf16* p32, int lane) {
    const int kb2 = (lane < 16) ? 0 : 16;   // byte offset of this half's K base
    frag_cvt c;
    c.p.lo = *(const v4u*)((const char*)p32 + kb2);
    c.p.hi = *(const v4u*)((const char*)p32 + 32 + kb2);
    return c.v;
}

// ---------------------------------------------------------------------------
// Generic fp32-in/fp32-out GEMM via bf16 WMMA. C = act(A @ W + bias).
// Block 64x64, 8 waves, wave = 16 rows x 32 cols. A and B tiles stored in
// lane-contiguous fragment layout (B transposed at fill) -> b128 frag loads.
// ---------------------------------------------------------------------------
template <int ACT>
__global__ __launch_bounds__(256)
void gemm_bf16_kernel(const float* __restrict__ A, int a_batch, int a_row,
                      const float* __restrict__ W, int w_batch,
                      const float* __restrict__ bias, int b_batch,
                      float* __restrict__ C, int c_batch, int c_row,
                      int M, int N, int Kd)
{
    __shared__ __bf16 As[64][40];       // row-major, 32 K contiguous, 80B stride
    __shared__ __bf16 Bs[4][16][40];    // [n-tile][col][K] transposed slabs

    const int tid  = threadIdx.x;
    const int lane = tid & 31;
    const int wave = tid >> 5;
    const int wr   = wave >> 1;         // 0..3 -> row block of 16
    const int wc   = wave & 1;          // 0..1 -> col block of 32
    const int m0   = blockIdx.y * 64;
    const int n0   = blockIdx.x * 64;
    const int bz   = blockIdx.z;

    const float* Ab = A + (long)bz * a_batch;
    const float* Wb = W + (long)bz * w_batch;
    const float* Bb = bias + (long)bz * b_batch;
    float*       Cb = C + (long)bz * c_batch;

    v8f acc0 = {0.f, 0.f, 0.f, 0.f, 0.f, 0.f, 0.f, 0.f};
    v8f acc1 = acc0;

    for (int kk = 0; kk < Kd; kk += 32) {
        // A tile: 64 rows x 16 K-pairs; float2 load -> packed b32 LDS store
        for (int e = tid; e < 64 * 16; e += 256) {
            int r = e >> 4, cp = e & 15;
            int gm = m0 + r, gk = kk + cp * 2;
            float2 v = make_float2(0.f, 0.f);
            if (gm < M) v = *(const float2*)(Ab + gm * a_row + gk);
            bfpair p; p.h[0] = (__bf16)v.x; p.h[1] = (__bf16)v.y;
            *(unsigned int*)&As[r][cp * 2] = p.u;
        }
        // B tile transposed: 16 K-pairs x 64 cols -> Bs[col-tile][col][K]
        for (int e = tid; e < 16 * 64; e += 256) {
            int rp = e >> 6, c = e & 63;
            int gk = kk + rp * 2, gn = n0 + c;
            float v0 = 0.f, v1 = 0.f;
            if (gn < N) { v0 = Wb[gk * N + gn]; v1 = Wb[(gk + 1) * N + gn]; }
            bfpair p; p.h[0] = (__bf16)v0; p.h[1] = (__bf16)v1;
            *(unsigned int*)&Bs[c >> 4][c & 15][rp * 2] = p.u;
        }
        if (kk + 32 < Kd)
            __builtin_prefetch(Wb + (kk + 32 + (tid >> 4)) * N + n0 + ((tid & 15) << 2), 0, 1);
        __syncthreads();
        v16bf a  = frag_ld32(&As[wr * 16 + (lane & 15)][0], lane);
        v16bf b0 = frag_ld32(&Bs[wc * 2 + 0][lane & 15][0], lane);
        v16bf b1 = frag_ld32(&Bs[wc * 2 + 1][lane & 15][0], lane);
        acc0 = __builtin_amdgcn_wmma_f32_16x16x32_bf16(false, a, false, b0, (short)0, acc0, false, false);
        // second of an identical pair shares A registers -> set reuse-A hint
        acc1 = __builtin_amdgcn_wmma_f32_16x16x32_bf16(false, a, false, b1, (short)0, acc1, true, false);
        __syncthreads();
    }

    // C/D layout: lane holds column n=l&15; VGPR r -> row r (+8 for lanes>=16)
    const int mbase = m0 + wr * 16 + ((lane < 16) ? 0 : 8);
#pragma unroll
    for (int t = 0; t < 2; ++t) {
        v8f acc = t ? acc1 : acc0;
        int gn = n0 + wc * 32 + t * 16 + (lane & 15);
        if (gn < N) {
            float bv = Bb[gn];
#pragma unroll
            for (int r = 0; r < 8; ++r) {
                int gm = mbase + r;
                if (gm < M) {
                    float v = acc[r] + bv;
                    if (ACT == 1) v = gelu_f(v);
                    Cb[gm * c_row + gn] = v;
                }
            }
        }
    }
}

// ---------------------------------------------------------------------------
// One-time swizzle: Wr1 (K, Kd, N) fp32 row-major -> bf16 [k][col][Kd]
// (column-contiguous) so the fused kernel can async-copy raw bytes to LDS.
// ---------------------------------------------------------------------------
__global__ void swizzle_wr1_kernel(const float* __restrict__ Wr1, __bf16* __restrict__ out)
{
    int i = blockIdx.x * blockDim.x + threadIdx.x;     // KM*HD*(HD/2) items
    if (i < KM * HD * (HD / 2)) {
        int j   = i & (HD - 1);                        // column (coalesced reads)
        int kdp = (i >> 9) & (HD / 2 - 1);             // K pair
        int k   = i >> 17;
        int kd  = kdp * 2;
        const float* src = Wr1 + (k * HD + kd) * HD + j;
        bfpair p; p.h[0] = (__bf16)src[0]; p.h[1] = (__bf16)src[HD];
        *(unsigned int*)&out[(k * HD + j) * HD + kd] = p.u;
    }
}

// ---------------------------------------------------------------------------
// Fused refine pipeline per mode k (blockIdx.z), 32 (b,t) rows per block:
//   tf = GELU(coarse @ Wt + bt) + attn_out      (LDS, bf16, packed stores)
//   ln = LayerNorm(tf)*g + b                    (in-place)
//   r  = GELU(ln @ Wr1[k] + br1[k])             (WMMA; B slabs double-buffered
//                                                via global_load_async_to_lds_b128)
//   off = r @ Wr2[k]                            (in-register contraction)
//   pred = coarse + off + br2[k]
// ---------------------------------------------------------------------------
__global__ __launch_bounds__(256)
void refine_fused_kernel(const float* __restrict__ coarse, const float* __restrict__ attn,
                         const float* __restrict__ Wt,  const float* __restrict__ bt,
                         const float* __restrict__ lng, const float* __restrict__ lnb,
                         const __bf16* __restrict__ Wbf, const float* __restrict__ br1,
                         const float* __restrict__ Wr2, const float* __restrict__ br2,
                         float* __restrict__ pred)
{
    __shared__ __bf16 Atile[32][520];     // 32 rows x 512, 1040B stride (16B aligned)
    __shared__ __bf16 Bs[2][8][16][40];   // double-buffered [nt][col][32 K contiguous]
    __shared__ float  mrow[32], rrow[32];
    __shared__ float  offs[32][2];
    __shared__ float  cc[32][2];

    const int tid  = threadIdx.x;
    const int lane = tid & 31;
    const int wave = tid >> 5;
    const int k    = blockIdx.z;
    const int r0   = blockIdx.x * 32;

    if (tid < 64) {
        int row = tid >> 1, o = tid & 1;
        int grow = r0 + row;
        int b = grow / TS, t = grow - b * TS;
        cc[row][o]   = coarse[(((b * KM) + k) * TS + t) * OD + o];
        offs[row][o] = 0.f;
    }
    __syncthreads();

    // Phase 1: traj_feats + attn broadcast (pairwise, packed b32 LDS stores)
    for (int e = tid; e < 32 * (HD / 2); e += 256) {
        int row = e >> 8, h = (e & 255) * 2;
        int b = (r0 + row) / TS;
        float c0 = cc[row][0], c1 = cc[row][1];
        float2 w0 = *(const float2*)(Wt + h);
        float2 w1 = *(const float2*)(Wt + HD + h);
        float2 bb = *(const float2*)(bt + h);
        float2 aa = *(const float2*)(attn + b * HD + h);
        bfpair p;
        p.h[0] = (__bf16)(gelu_f(c0 * w0.x + c1 * w1.x + bb.x) + aa.x);
        p.h[1] = (__bf16)(gelu_f(c0 * w0.y + c1 * w1.y + bb.y) + aa.y);
        *(unsigned int*)&Atile[row][h] = p.u;
    }
    __syncthreads();

    // Phase 2: per-row LayerNorm stats (each wave owns 4 rows)
    for (int rr = 0; rr < 4; ++rr) {
        int row = wave * 4 + rr;
        float s = 0.f, sq = 0.f;
        for (int jp = lane; jp < HD / 2; jp += 32) {
            bfpair p; p.u = *(const unsigned int*)&Atile[row][jp * 2];
            float x0 = (float)p.h[0], x1 = (float)p.h[1];
            s += x0 + x1; sq += x0 * x0 + x1 * x1;
        }
#pragma unroll
        for (int d = 16; d >= 1; d >>= 1) {
            s  += __shfl_xor(s, d, 32);
            sq += __shfl_xor(sq, d, 32);
        }
        if (lane == 0) {
            float m = s * (1.0f / HD);
            float v = sq * (1.0f / HD) - m * m;
            mrow[row] = m;
            rrow[row] = rsqrtf(v + 1e-5f);
        }
    }
    __syncthreads();

    for (int e = tid; e < 32 * (HD / 2); e += 256) {
        int row = e >> 8, h = (e & 255) * 2;
        float m = mrow[row], r = rrow[row];
        float2 g2 = *(const float2*)(lng + h);
        float2 b2 = *(const float2*)(lnb + h);
        bfpair p; p.u = *(const unsigned int*)&Atile[row][h];
        bfpair q;
        q.h[0] = (__bf16)(((float)p.h[0] - m) * r * g2.x + b2.x);
        q.h[1] = (__bf16)(((float)p.h[1] - m) * r * g2.y + b2.y);
        *(unsigned int*)&Atile[row][h] = q.u;
    }
    __syncthreads();

    // Phase 3: ln @ Wr1 via WMMA; B slabs async-copied, double-buffered so the
    // DMA for slab i+1 overlaps the WMMAs on slab i.
    const int wr = wave >> 2;   // 0..1 -> row block of 16
    const int wc = wave & 3;    // 0..3 -> col block of 32 within 128-col chunk
    const __bf16* Wk = Wbf + (long)k * HD * HD;

    float offa[16];
#pragma unroll
    for (int i = 0; i < 16; ++i) offa[i] = 0.f;

    for (int nc = 0; nc < HD; nc += 128) {
        // issue one 8KB slab (8 nt x 16 col x 64B): 2 async b128 per thread
        auto issue_slab = [&](int kk, int buf) {
#pragma unroll
            for (int it = 0; it < 2; ++it) {
                int e   = tid + it * 256;
                int nt  = e >> 6, n = (e >> 2) & 15, sub = e & 3;
                int col = nc + nt * 16 + n;
                const __bf16* g = Wk + col * HD + kk + sub * 8;
                unsigned lds = (unsigned)(unsigned long long)&Bs[buf][nt][n][sub * 8];
                asm volatile("global_load_async_to_lds_b128 %0, %1, off"
                             :: "v"(lds), "v"(g) : "memory");
            }
        };

        v8f acc0 = {0.f, 0.f, 0.f, 0.f, 0.f, 0.f, 0.f, 0.f};
        v8f acc1 = acc0;

        issue_slab(0, 0);
        for (int kk = 0, i = 0; kk < HD; kk += 32, ++i) {
            const int cur = i & 1;
            if (kk + 32 < HD) {
                issue_slab(kk + 32, cur ^ 1);
                // 2 per-wave async ops per slab, in-order retire: <=2 pending
                // means the *older* slab has fully landed in LDS.
                asm volatile("s_wait_asynccnt 0x2" ::: "memory");
            } else {
                asm volatile("s_wait_asynccnt 0x0" ::: "memory");
            }
            __syncthreads();   // slab `cur` visible to all waves
            v16bf a  = frag_ld32(&Atile[wr * 16 + (lane & 15)][kk], lane);
            v16bf b0 = frag_ld32(&Bs[cur][wc * 2 + 0][lane & 15][0], lane);
            v16bf b1 = frag_ld32(&Bs[cur][wc * 2 + 1][lane & 15][0], lane);
            acc0 = __builtin_amdgcn_wmma_f32_16x16x32_bf16(false, a, false, b0, (short)0, acc0, false, false);
            acc1 = __builtin_amdgcn_wmma_f32_16x16x32_bf16(false, a, false, b1, (short)0, acc1, true, false);
            __syncthreads();   // all reads of slab `cur` done before reuse at i+2
        }

        // GELU + contraction with Wr2 (never materialize r)
#pragma unroll
        for (int t = 0; t < 2; ++t) {
            v8f acc = t ? acc1 : acc0;
            int f = nc + wc * 32 + t * 16 + (lane & 15);
            float bv = br1[k * HD + f];
            float w0 = Wr2[(k * HD + f) * 2 + 0];
            float w1 = Wr2[(k * HD + f) * 2 + 1];
#pragma unroll
            for (int r = 0; r < 8; ++r) {
                float v = gelu_f(acc[r] + bv);
                offa[r * 2 + 0] += v * w0;
                offa[r * 2 + 1] += v * w1;
            }
        }
    }

    // reduce across the 16 columns held by each half-wave
#pragma unroll
    for (int i = 0; i < 16; ++i) {
#pragma unroll
        for (int d = 8; d >= 1; d >>= 1) offa[i] += __shfl_xor(offa[i], d, 16);
    }
    if ((lane & 15) == 0) {
        int rbase = wr * 16 + ((lane < 16) ? 0 : 8);
#pragma unroll
        for (int r = 0; r < 8; ++r) {
            atomicAdd(&offs[rbase + r][0], offa[r * 2 + 0]);
            atomicAdd(&offs[rbase + r][1], offa[r * 2 + 1]);
        }
    }
    __syncthreads();

    if (tid < 64) {
        int row = tid >> 1, o = tid & 1;
        int grow = r0 + row;
        int b = grow / TS, t = grow - b * TS;
        pred[(((b * KM) + k) * TS + t) * OD + o] = cc[row][o] + offs[row][o] + br2[k * 2 + o];
    }
}

// ---------------------------------------------------------------------------
// Small elementwise / reduction kernels
// ---------------------------------------------------------------------------
__global__ void pack_combined_kernel(const float* __restrict__ a, const float* __restrict__ b,
                                     const float* __restrict__ c, float* __restrict__ out)
{
    int i = blockIdx.x * blockDim.x + threadIdx.x;
    if (i < NB * HD) {
        int row = i / HD, h = i - row * HD;
        float* o = out + row * 3 * HD;
        o[h] = a[i]; o[HD + h] = b[i]; o[2 * HD + h] = c[i];
    }
}

__global__ void mode_features_kernel(const float* __restrict__ ctx, const float* __restrict__ mq,
                                     float* __restrict__ mf)
{
    int i = blockIdx.x * blockDim.x + threadIdx.x;
    if (i < NB * KM * HD) {
        int h = i % HD;
        int k = (i / HD) % KM;
        int b = i / (HD * KM);
        mf[i] = ctx[b * HD + h] + mq[k * HD + h];
    }
}

__global__ __launch_bounds__(64)
void conf_kernel(const float* __restrict__ mf, const float* __restrict__ Wf1,
                 const float* __restrict__ bf1, const float* __restrict__ Wf2,
                 const float* __restrict__ bf2, float* __restrict__ conf)
{
    __shared__ float part[2];
    int row = blockIdx.x;           // row = b*KM + k
    int tid = threadIdx.x;          // 64 columns of Wf1
    const float* a = mf + row * HD;
    float s = 0.f;
    for (int i = 0; i < HD; ++i) s += a[i] * Wf1[i * 64 + tid];
    s = fmaxf(s + bf1[tid], 0.f) * Wf2[tid];
#pragma unroll
    for (int d = 16; d >= 1; d >>= 1) s += __shfl_xor(s, d, 32);
    if ((tid & 31) == 0) part[tid >> 5] = s;
    __syncthreads();
    if (tid == 0) conf[row] = part[0] + part[1] + bf2[0];
}

__global__ void softmax_kernel(const float* __restrict__ conf, float* __restrict__ probs)
{
    int b = blockIdx.x;
    if (threadIdx.x == 0) {
        float mx = -1e30f;
        for (int k = 0; k < KM; ++k) mx = fmaxf(mx, conf[b * KM + k]);
        float e[KM]; float s = 0.f;
        for (int k = 0; k < KM; ++k) { e[k] = expf(conf[b * KM + k] - mx); s += e[k]; }
        float inv = 1.0f / s;
        for (int k = 0; k < KM; ++k) probs[b * KM + k] = e[k] * inv;
    }
}

// ---------------------------------------------------------------------------
extern "C" void kernel_launch(void* const* d_in, const int* in_sizes, int n_in,
                              void* d_out, int out_size, void* d_ws, size_t ws_size,
                              hipStream_t stream)
{
    const float* interaction = (const float*)d_in[0];
    const float* cooperative = (const float*)d_in[1];
    const float* map_ctx     = (const float*)d_in[2];
    const float* mode_q      = (const float*)d_in[3];
    const float* Wc1 = (const float*)d_in[4];  const float* bc1 = (const float*)d_in[5];
    const float* Wc2 = (const float*)d_in[6];  const float* bc2 = (const float*)d_in[7];
    const float* Wk1 = (const float*)d_in[8];  const float* bk1 = (const float*)d_in[9];
    const float* Wk2 = (const float*)d_in[10]; const float* bk2 = (const float*)d_in[11];
    const float* Wt  = (const float*)d_in[12]; const float* bt  = (const float*)d_in[13];
    // Wq_a (14/15), Wk_a (16/17): dead — softmax is exactly uniform in reference
    const float* Wv  = (const float*)d_in[18]; const float* bv  = (const float*)d_in[19];
    const float* Wo  = (const float*)d_in[20]; const float* bo  = (const float*)d_in[21];
    const float* lng = (const float*)d_in[22]; const float* lnb = (const float*)d_in[23];
    const float* Wr1 = (const float*)d_in[24]; const float* br1 = (const float*)d_in[25];
    const float* Wr2 = (const float*)d_in[26]; const float* br2 = (const float*)d_in[27];
    const float* Wf1 = (const float*)d_in[28]; const float* bf1 = (const float*)d_in[29];
    const float* Wf2 = (const float*)d_in[30]; const float* bf2 = (const float*)d_in[31];

    float* ws = (float*)d_ws;
    float* combined = ws;                          // 512*1536
    float* tmp1     = combined + NB * 3 * HD;      // 512*512
    float* context  = tmp1 + NB * HD;              // 512*512
    float* vbuf     = context + NB * HD;           // 512*512
    float* attn     = vbuf + NB * HD;              // 512*512
    float* mf       = attn + NB * HD;              // 512*6*512
    float* hbuf     = mf + NB * KM * HD;           // 512*6*1024
    float* coarse   = hbuf + NB * KM * 2 * HD;     // 512*6*120
    float* confr    = coarse + NB * KM * TS * OD;  // 3072
    __bf16* wr1bf   = (__bf16*)(confr + NB * KM);  // 6*512*512 bf16 (3MB)

    float* pred  = (float*)d_out;                  // 512*6*60*2
    float* probs = pred + NB * KM * TS * OD;       // 512*6

    dim3 blk(256);

    // 0) one-shot bf16 swizzle of Wr1 into column-contiguous layout
    swizzle_wr1_kernel<<<(KM * HD * (HD / 2) + 255) / 256, blk, 0, stream>>>(Wr1, wr1bf);

    // 1) concat context inputs
    pack_combined_kernel<<<(NB * HD + 255) / 256, blk, 0, stream>>>(interaction, cooperative, map_ctx, combined);

    // 2) context MLP
    gemm_bf16_kernel<1><<<dim3(8, 8, 1), blk, 0, stream>>>(combined, 0, 3 * HD, Wc1, 0, bc1, 0,
                                                           tmp1, 0, HD, NB, HD, 3 * HD);
    gemm_bf16_kernel<0><<<dim3(8, 8, 1), blk, 0, stream>>>(tmp1, 0, HD, Wc2, 0, bc2, 0,
                                                           context, 0, HD, NB, HD, HD);

    // 3) collapsed attention: attn_out = (context @ Wv + bv) @ Wo + bo
    gemm_bf16_kernel<0><<<dim3(8, 8, 1), blk, 0, stream>>>(context, 0, HD, Wv, 0, bv, 0,
                                                           vbuf, 0, HD, NB, HD, HD);
    gemm_bf16_kernel<0><<<dim3(8, 8, 1), blk, 0, stream>>>(vbuf, 0, HD, Wo, 0, bo, 0,
                                                           attn, 0, HD, NB, HD, HD);

    // 4) mode_features = mode_queries + context
    mode_features_kernel<<<(NB * KM * HD + 255) / 256, blk, 0, stream>>>(context, mode_q, mf);

    // 5) coarse heads, grouped GEMM over modes (blockIdx.z = k)
    gemm_bf16_kernel<1><<<dim3(16, 8, KM), blk, 0, stream>>>(mf, HD, KM * HD, Wk1, HD * 2 * HD, bk1, 2 * HD,
                                                             hbuf, 2 * HD, KM * 2 * HD, NB, 2 * HD, HD);
    gemm_bf16_kernel<0><<<dim3(2, 8, KM), blk, 0, stream>>>(hbuf, 2 * HD, KM * 2 * HD, Wk2, 2 * HD * TS * OD,
                                                            bk2, TS * OD, coarse, TS * OD, KM * TS * OD,
                                                            NB, TS * OD, 2 * HD);

    // 6) fused refine pipeline (dominant 96.6 GFLOP)
    refine_fused_kernel<<<dim3((NB * TS) / 32, 1, KM), blk, 0, stream>>>(
        coarse, attn, Wt, bt, lng, lnb, wr1bf, br1, Wr2, br2, pred);

    // 7) confidence head + softmax over modes
    conf_kernel<<<dim3(NB * KM), dim3(64), 0, stream>>>(mf, Wf1, bf1, Wf2, bf2, confr);
    softmax_kernel<<<dim3(NB), dim3(32), 0, stream>>>(confr, probs);
}
